// SpectralCirculantLayer_31542239821935
// MI455X (gfx1250) — matrix operand order repaired and checked by reading, not compiled, not run on hardware
//
#include <hip/hip_runtime.h>
#include <hip/hip_bf16.h>

#define NPAD   4096
#define KDIM   4096
#define BATCH  8192

#define BM 128
#define BN 128
#define BK 32
#define LDP 40           // padded LDS row pitch in ushorts (80 B) -> conflict-free b128 reads
#define TILE_USHORTS (BM * LDP)   // 5120 ushorts = 10240 B per tile buffer

typedef __attribute__((ext_vector_type(16))) __bf16       bf16x16;
typedef __attribute__((ext_vector_type(8)))  float        f32x8;
typedef __attribute__((ext_vector_type(4)))  float        f32x4;
typedef __attribute__((ext_vector_type(4)))  unsigned int u32x4;
typedef __attribute__((ext_vector_type(4)))  int          i32x4;
typedef __attribute__((ext_vector_type(8)))  int          i32x8;

union FragU {
    u32x4   q[2];
    bf16x16 v;
};

__device__ __forceinline__ unsigned short f2bf(float f) {
    unsigned int u = __float_as_uint(f);
    unsigned int r = u + 0x7FFFu + ((u >> 16) & 1u);   // round-to-nearest-even
    return (unsigned short)(r >> 16);
}

// -------------------------------------------------------------------------
// TDM: 2-D tile load (32 cols x 128 rows of bf16, row stride 4096 elems)
// into LDS at lds_off; DMA-side padding of 4 DWORDs after every 16 DWORDs
// reproduces the 80 B LDS row pitch ([128][LDP] ushort layout).
// D# packing per CDNA5 ISA ch.8.
// -------------------------------------------------------------------------
__device__ __forceinline__ void tdm_load_tile(unsigned lds_off,
                                              const unsigned short* gptr,
                                              unsigned tensor_rows) {
    unsigned long long ga = (unsigned long long)(size_t)gptr;
    u32x4 g0;
    g0[0] = 1u;                                            // count=1 (valid), user mode
    g0[1] = lds_off;                                       // lds_addr [63:32]
    g0[2] = (unsigned)(ga & 0xFFFFFFFFu);                  // global_addr lo
    g0[3] = (unsigned)((ga >> 32) & 0x1FFFFFFu)            // global_addr hi [120:96]
          | (2u << 30);                                    // type=2 (image)

    i32x8 g1;
    g1[0] = (1 << 16)        // data_size = 2 bytes
          | (1 << 20)        // pad_enable
          | (3 << 22)        // pad_interval: 16 DWORDs (one 64 B tile row)
          | (3 << 25);       // pad_amount:   4 DWORDs (16 B)
    g1[1] = (int)(4096u << 16);                            // tensor_dim0 = 4096 (lo16)
    g1[2] = (int)((tensor_rows & 0xFFFFu) << 16);          // td0 hi=0 | tensor_dim1 lo
    g1[3] = (int)((tensor_rows >> 16) | (32u << 16));      // tensor_dim1 hi | tile_dim0=32
    g1[4] = 128;                                           // tile_dim1=128, tile_dim2=0
    g1[5] = 4096;                                          // tensor_dim0_stride lo (elems)
    g1[6] = 0;
    g1[7] = 0;

    i32x4 z4 = {0, 0, 0, 0};
#if __clang_major__ >= 23
    i32x8 z8 = {0, 0, 0, 0, 0, 0, 0, 0};
    __builtin_amdgcn_tensor_load_to_lds(g0, g1, z4, z4, z8, 0);
#else
    __builtin_amdgcn_tensor_load_to_lds(g0, g1, z4, z4, 0);
#endif
}

// -------------------------------------------------------------------------
// Kernel 1: h[n] = real inverse DFT of the Hermitian half-spectrum.
// -------------------------------------------------------------------------
__global__ void build_h_kernel(const float* __restrict__ wr,
                               const float* __restrict__ wi,
                               float* __restrict__ h) {
    int n = blockIdx.x * blockDim.x + threadIdx.x;     // 0..4095
    const float step = 6.2831853071795864769f / 4096.0f;
    float acc = 0.0f;
    for (int k = 1; k < 2048; ++k) {
        int ph = (k * n) & 4095;
        float s, c;
        __sincosf(step * (float)ph, &s, &c);
        acc = fmaf(wr[k], c, acc);
        acc = fmaf(-wi[k], s, acc);
    }
    float v = wr[0] + ((n & 1) ? -wr[2048] : wr[2048]) + 2.0f * acc;
    h[n] = v * (1.0f / 4096.0f);
}

// -------------------------------------------------------------------------
// Kernel 2: Bt[n][k] = bf16( h[(n-k) & 4095] )
// -------------------------------------------------------------------------
__global__ void build_bt_kernel(const float* __restrict__ h,
                                unsigned short* __restrict__ bt) {
    int n = blockIdx.x;
    unsigned short* row = bt + (size_t)n * KDIM;
    for (int k = threadIdx.x; k < KDIM; k += blockDim.x)
        row[k] = f2bf(h[(n - k) & 4095]);
}

// -------------------------------------------------------------------------
// Kernel 3: x f32 -> bf16
// -------------------------------------------------------------------------
__global__ void convert_x_kernel(const float* __restrict__ x,
                                 unsigned short* __restrict__ xb) {
    size_t base = ((size_t)blockIdx.x * blockDim.x + threadIdx.x) * 8;
    f32x4 a = *(const f32x4*)(x + base);
    f32x4 b = *(const f32x4*)(x + base + 4);
    unsigned short t[8];
    for (int i = 0; i < 4; ++i) t[i]     = f2bf(a[i]);
    for (int i = 0; i < 4; ++i) t[i + 4] = f2bf(b[i]);
    *(u32x4*)(xb + base) = *(const u32x4*)t;
}

// -------------------------------------------------------------------------
// Kernel 4: Y = Xb * Bt^T + bias.  bf16 WMMA, 3-stage TDM pipeline.
// Block tile 128x128, 8 waves (2 in M x 4 in N), wave tile 64x32 = 4x2 WMMA.
// -------------------------------------------------------------------------
__global__ __launch_bounds__(256) void spectral_gemm_kernel(
        const unsigned short* __restrict__ Xb,
        const unsigned short* __restrict__ Bt,
        const float* __restrict__ bias,
        float* __restrict__ Y) {

    __shared__ unsigned short As[3][BM][LDP];   // 3 x 10240 B
    __shared__ unsigned short Bs[3][BN][LDP];   // 3 x 10240 B   (60 KB total)

    const int tid  = threadIdx.x;
    const int lane = tid & 31;
    const int wid  = tid >> 5;
    const int wm   = wid & 1;          // 2 waves across M
    const int wn   = wid >> 1;         // 4 waves across N
    const int n0   = blockIdx.x * BN;
    const int m0   = blockIdx.y * BM;

    const unsigned short* ag = Xb + (size_t)m0 * KDIM;   // A tile row base
    const unsigned short* bg = Bt + (size_t)n0 * KDIM;   // B tile row base

    const unsigned asBase = (unsigned)(size_t)&As[0][0][0];
    const unsigned bsBase = (unsigned)(size_t)&Bs[0][0][0];
    const unsigned tileBytes = TILE_USHORTS * 2u;         // 10240

    f32x8 acc[4][2];
    for (int ms = 0; ms < 4; ++ms)
        for (int ns = 0; ns < 2; ++ns)
            for (int r = 0; r < 8; ++r)
                acc[ms][ns][r] = 0.0f;

    const int KSTEPS = KDIM / BK;   // 128

    // ---- prologue: DMA k-tiles 0 and 1 into buffers 0 and 1 (wave 0 drives TDM)
    if (wid == 0) {
        tdm_load_tile(asBase,             ag,      BATCH);
        tdm_load_tile(bsBase,             bg,      NPAD);
        tdm_load_tile(asBase + tileBytes, ag + BK, BATCH);
        tdm_load_tile(bsBase + tileBytes, bg + BK, NPAD);
        __builtin_amdgcn_s_wait_tensorcnt(2);   // tile 0 landed; tile 1 may fly
    }
    __syncthreads();

    int buf = 0;           // buffer holding k-tile kt
    #pragma unroll 1
    for (int kt = 0; kt < KSTEPS; ++kt) {
        const int nxt2 = (buf + 2 >= 3) ? (buf - 1) : (buf + 2);   // (buf+2)%3
        const bool issue = (kt + 2) < KSTEPS;

        // ---- kick the k+2 tile DMA into the third buffer (flies across 2 iters)
        if (issue && wid == 0) {
            const int kn = (kt + 2) * BK;
            tdm_load_tile(asBase + (unsigned)nxt2 * tileBytes, ag + kn, BATCH);
            tdm_load_tile(bsBase + (unsigned)nxt2 * tileBytes, bg + kn, NPAD);
        }

        const unsigned short (*Asb)[LDP] = As[buf];
        const unsigned short (*Bsb)[LDP] = Bs[buf];

        // ---- B fragments: lane<16 -> col N=lane, K 0..15 ; lane>=16 -> K 16..31
        FragU bfrag[2];
        {
            const int kb = (lane < 16) ? 0 : 16;
            for (int ns = 0; ns < 2; ++ns) {
                const int nr = wn * 32 + ns * 16 + (lane & 15);
                bfrag[ns].q[0] = *(const u32x4*)&Bsb[nr][kb];
                bfrag[ns].q[1] = *(const u32x4*)&Bsb[nr][kb + 8];
            }
        }

        // ---- A fragments + WMMA
        // lane<16: row M=lane, K 0..15; lane>=16: K 16..31
        const int ka = (lane < 16) ? 0 : 16;
        for (int ms = 0; ms < 4; ++ms) {
            const int mr = wm * 64 + ms * 16 + (lane & 15);
            FragU afrag;
            afrag.q[0] = *(const u32x4*)&Asb[mr][ka];
            afrag.q[1] = *(const u32x4*)&Asb[mr][ka + 8];
            for (int ns = 0; ns < 2; ++ns) {
                acc[ms][ns] = __builtin_amdgcn_wmma_f32_16x16x32_bf16(
                    /*neg_a=*/false, afrag.v,
                    /*neg_b=*/false, bfrag[ns].v,
                    /*c_mod=*/(short)0, acc[ms][ns],
                    /*reuse_a=*/false, /*reuse_b=*/false);
            }
        }

        // ---- guarantee k+1's buffer has landed; leave k+2's DMA in flight
        if (wid == 0) {
            if (issue) __builtin_amdgcn_s_wait_tensorcnt(2);
            else       __builtin_amdgcn_s_wait_tensorcnt(0);
        }
        __syncthreads();

        buf = (buf + 1 >= 3) ? 0 : (buf + 1);
    }

    // ---- epilogue: C layout (VGPR r, lane L): M = base + r + (L>=16 ? 8:0), N = base + (L&15)
    for (int ms = 0; ms < 4; ++ms) {
        const int mrow = m0 + wm * 64 + ms * 16 + ((lane >= 16) ? 8 : 0);
        for (int ns = 0; ns < 2; ++ns) {
            const int ncol = n0 + wn * 32 + ns * 16 + (lane & 15);
            const float bv = bias[ncol];
            float* yp = Y + (size_t)mrow * NPAD + ncol;
            for (int r = 0; r < 8; ++r)
                yp[(size_t)r * NPAD] = acc[ms][ns][r] + bv;
        }
    }
}

// -------------------------------------------------------------------------
extern "C" void kernel_launch(void* const* d_in, const int* in_sizes, int n_in,
                              void* d_out, int out_size, void* d_ws, size_t ws_size,
                              hipStream_t stream) {
    (void)in_sizes; (void)n_in; (void)out_size; (void)ws_size;

    const float* x    = (const float*)d_in[0];   // (8192, 4096) f32
    const float* wr   = (const float*)d_in[1];   // (2049,) f32
    const float* wi   = (const float*)d_in[2];   // (2049,) f32
    const float* bias = (const float*)d_in[3];   // (4096,) f32
    float*       out  = (float*)d_out;           // (8192, 4096) f32

    // ws layout: [ Xb bf16 64 MB ][ Bt bf16 32 MB ][ h f32 16 KB ]
    unsigned short* Xb = (unsigned short*)d_ws;
    unsigned short* Bt = Xb + (size_t)BATCH * KDIM;
    float*          h  = (float*)(Bt + (size_t)NPAD * KDIM);

    build_h_kernel<<<NPAD / 256, 256, 0, stream>>>(wr, wi, h);
    build_bt_kernel<<<NPAD, 256, 0, stream>>>(h, Bt);
    convert_x_kernel<<<(int)(((size_t)BATCH * KDIM / 8) / 256), 256, 0, stream>>>(x, Xb);

    dim3 grid(NPAD / BN, BATCH / BM);
    spectral_gemm_kernel<<<grid, 256, 0, stream>>>(Xb, Bt, bias, out);
}
